// EQ_GNN_20023137534500
// MI455X (gfx1250) — matrix operand order repaired and checked by reading, not compiled, not run on hardware
//
#include <hip/hip_runtime.h>
#include <hip/hip_bf16.h>
#include <math.h>

typedef __attribute__((ext_vector_type(16))) _Float16 v16h;
typedef __attribute__((ext_vector_type(8)))  _Float16 v8h;
typedef __attribute__((ext_vector_type(8)))  float    v8f;

#define N_PART 128
#define F_DIM  32
#define H_DIM  32
#define EPN    127                 /* edges per node */
#define EPB    (N_PART*(N_PART-1)) /* edges per batch = 16256 */

// d_ws layout (bytes):
//   [0, 14336)                 : 14 packed f16 weight B-fragments (14 x 32 lanes x 16 halves)
//   [14336, 14336+1048576)     : h packed to f16 (16384 x 32)
//   [1062912, 1062912+2097152) : m_i sums, f32 (16384 x 32)
#define WP_BYTES  14336
#define H16_BYTES 1048576

__device__ __forceinline__ float sigmoid_f(float v){ return 1.0f/(1.0f+__expf(-v)); }
__device__ __forceinline__ float silu_f(float v){ return v*sigmoid_f(v); }

__device__ __forceinline__ v8f wmma_f16(v16h a, v16h b, v8f c){
  return __builtin_amdgcn_wmma_f32_16x16x32_f16(false, a, false, b, (short)0, c, false, false);
}

__device__ __forceinline__ v16h load_frag(const _Float16* __restrict__ wp, int f, int lane){
  return *(const v16h*)(wp + f*512 + lane*16);
}

// A fragment (16-bit A 16x32 layout) from packed-f16 row: two 16B loads.
__device__ __forceinline__ v16h afrag_f16row(const _Float16* __restrict__ row, int kh){
  v8h lo = *(const v8h*)(row + kh);
  v8h hi = *(const v8h*)(row + 16 + kh);
  return __builtin_shufflevector(lo, hi, 0,1,2,3,4,5,6,7,8,9,10,11,12,13,14,15);
}

// A fragment gathered from the per-wave f32 LDS tile (C->A relayout).
__device__ __forceinline__ v16h afrag_from_T(const float T[16][32], int m, int kh){
  v16h a;
#pragma unroll
  for (int q = 0; q < 8; ++q){
    a[q]   = (_Float16)T[m][kh + q];
    a[8+q] = (_Float16)T[m][16 + kh + q];
  }
  return a;
}

// ---------------------------------------------------------------------------
// Pack kernel A: 14 weight B-fragments (32x16 f16 each) in per-lane layout.
// ---------------------------------------------------------------------------
__global__ __launch_bounds__(448) void pack_weights_kernel(
    const float* __restrict__ ew1, const float* __restrict__ ew2,
    const float* __restrict__ cw1, const float* __restrict__ nw1,
    const float* __restrict__ nw2, _Float16* __restrict__ wp)
{
  const int t = threadIdx.x;           // 0..447
  const int f = t >> 5;
  const int lane = t & 31;
  const float* W; int k0, n0;
  switch (f) {
    case  0: W = ew1; k0 =  0; n0 =  0; break;
    case  1: W = ew1; k0 = 32; n0 =  0; break;
    case  2: W = ew1; k0 =  0; n0 = 16; break;
    case  3: W = ew1; k0 = 32; n0 = 16; break;
    case  4: W = ew2; k0 =  0; n0 =  0; break;
    case  5: W = ew2; k0 =  0; n0 = 16; break;
    case  6: W = cw1; k0 =  0; n0 =  0; break;
    case  7: W = cw1; k0 =  0; n0 = 16; break;
    case  8: W = nw1; k0 =  0; n0 =  0; break;
    case  9: W = nw1; k0 = 32; n0 =  0; break;
    case 10: W = nw1; k0 =  0; n0 = 16; break;
    case 11: W = nw1; k0 = 32; n0 = 16; break;
    case 12: W = nw2; k0 =  0; n0 =  0; break;
    default: W = nw2; k0 =  0; n0 = 16; break;
  }
  const int n  = n0 + (lane & 15);
  const int kb = k0 + ((lane >> 4) << 3);
  _Float16* dst = wp + f*512 + lane*16;
#pragma unroll
  for (int q = 0; q < 16; ++q) dst[q] = (_Float16)W[(kb + (q < 8 ? q : q + 8)) * H_DIM + n];
}

// Pack kernel B: h (f32) -> h16. 512 blocks x 256 threads x 4 elems.
__global__ __launch_bounds__(256) void pack_h_kernel(
    const float* __restrict__ h, _Float16* __restrict__ h16)
{
  const int idx = (blockIdx.x * 256 + threadIdx.x) * 4;
#pragma unroll
  for (int q = 0; q < 4; ++q) h16[idx + q] = (_Float16)h[idx + q];
}

// ---------------------------------------------------------------------------
// Kernel 1: one wave per (batch b, node i); loops the 8 edge tiles of node i.
// Weight fragments + hi + scalars loaded once; 64 WMMAs per wave; all node
// reductions carried in registers across tiles (no atomics, no barriers).
// Lanes l and l+16 hold the same edge, so per-edge LDS scalars are stored
// unconditionally from both halves (identical values) -- no exec-mask branch.
// ---------------------------------------------------------------------------
__global__ __launch_bounds__(32) void eq_gnn_edge_kernel(
    const float* __restrict__ x,   const _Float16* __restrict__ h16,
    const float* __restrict__ dsb,
    const float* __restrict__ ew1, const float* __restrict__ eb1,
    const float* __restrict__ eb2, const float* __restrict__ cb1,
    const float* __restrict__ cw2, const float* __restrict__ aw,
    const float* __restrict__ ab,  const _Float16* __restrict__ wp,
    float* __restrict__ x_out, float* __restrict__ mi_ws)
{
  __shared__ float T[16][32];
  __shared__ float sD2[16];
  __shared__ float sDS[16];
  __shared__ float sG[16];

  const int lane  = threadIdx.x;
  const int b     = blockIdx.x >> 7;
  const int i     = blockIdx.x & 127;
  const int m     = lane & 15;
  const int grp   = lane >> 4;
  const int mbase = grp * 8;
  const int kh    = grp * 8;
  const int n0    = m;

  // persistent packed weight fragments (8 x v16h)
  const v16h Be1_00 = load_frag(wp, 0, lane);  // e_w1 K=hi,  N tile 0
  const v16h Be1_10 = load_frag(wp, 1, lane);  // e_w1 K=hj,  N tile 0
  const v16h Be1_01 = load_frag(wp, 2, lane);  // e_w1 K=hi,  N tile 1
  const v16h Be1_11 = load_frag(wp, 3, lane);  // e_w1 K=hj,  N tile 1
  const v16h Be2_0  = load_frag(wp, 4, lane);
  const v16h Be2_1  = load_frag(wp, 5, lane);
  const v16h Bc1_0  = load_frag(wp, 6, lane);
  const v16h Bc1_1  = load_frag(wp, 7, lane);

  // hi row is shared by every edge of this block: load once
  const v16h a_hi = afrag_f16row(h16 + (b * N_PART + i) * F_DIM, kh);

  // rank-1 (d^2, ds^2) columns of e_w1 + biases, per-lane
  const float w64a = ew1[64*H_DIM + n0],      w65a = ew1[65*H_DIM + n0],      b1a = eb1[n0];
  const float w64b = ew1[64*H_DIM + n0 + 16], w65b = ew1[65*H_DIM + n0 + 16], b1b = eb1[n0 + 16];
  const float b2a  = eb2[n0],  b2b  = eb2[n0 + 16];
  const float cb1a = cb1[n0],  cb1b = cb1[n0 + 16];
  const float abv  = ab[0];
  float awr[16], c2r[16];
#pragma unroll
  for (int c = 0; c < 16; ++c) { awr[c] = aw[grp*16 + c]; c2r[c] = cw2[grp*16 + c]; }

  const float* xb = x + b * (N_PART * 3);
  const float xi0 = xb[i*3+0], xi1 = xb[i*3+1], xi2 = xb[i*3+2];
  const float* dsrow = dsb + b * EPB + i * EPN;

  float cs0 = 0.0f, cs1 = 0.0f;          // m_i column sums (cols n0, n0+16)
  float ux = 0.0f, uy = 0.0f, uz = 0.0f; // x_upd sums

  for (int t = 0; t < 8; ++t) {
    const int jp   = t * 16 + m;
    const bool valid = (jp < EPN);
    const int jpc  = valid ? jp : (EPN - 1);
    const int j    = jpc + (jpc >= i ? 1 : 0);

    // geometry (both lane halves compute identical values for their edge)
    const float r0 = xi0 - xb[j*3+0];
    const float r1 = xi1 - xb[j*3+1];
    const float r2 = xi2 - xb[j*3+2];
    const float d2 = r0*r0 + r1*r1 + r2*r2 + 1e-6f;
    const float d  = sqrtf(d2);
    const float dsv = dsrow[jpc];
    sD2[m] = d2;              // unconditional: halves write identical values
    sDS[m] = dsv * dsv;

    const v16h a_hj = afrag_f16row(h16 + (b * N_PART + j) * F_DIM, kh);
    __asm__ volatile("s_wait_dscnt 0" ::: "memory");

    // layer 1
    v8f acc0 = {}, acc1 = {};
    acc0 = wmma_f16(a_hi, Be1_00, acc0);
    acc0 = wmma_f16(a_hj, Be1_10, acc0);
    acc1 = wmma_f16(a_hi, Be1_01, acc1);
    acc1 = wmma_f16(a_hj, Be1_11, acc1);
#pragma unroll
    for (int v = 0; v < 8; ++v) {
      const float dd = sD2[mbase + v], ss = sDS[mbase + v];
      T[mbase+v][n0]      = silu_f(acc0[v] + dd*w64a + ss*w65a + b1a);
      T[mbase+v][n0 + 16] = silu_f(acc1[v] + dd*w64b + ss*w65b + b1b);
    }
    __asm__ volatile("s_wait_dscnt 0" ::: "memory");

    // layer 2 -> m0
    const v16h a2 = afrag_from_T(T, m, kh);
    v8f s0 = {}, s1 = {};
    s0 = wmma_f16(a2, Be2_0, s0);
    s1 = wmma_f16(a2, Be2_1, s1);
#pragma unroll
    for (int v = 0; v < 8; ++v) {
      const float q0 = silu_f(s0[v] + b2a);
      const float q1 = silu_f(s1[v] + b2b);
      s0[v] = q0; s1[v] = q1;
      T[mbase+v][n0] = q0; T[mbase+v][n0 + 16] = q1;
    }
    __asm__ volatile("s_wait_dscnt 0" ::: "memory");

    // gate g = sigmoid(m0 . a_w + a_b) per edge
    float part = 0.0f;
#pragma unroll
    for (int c = 0; c < 16; ++c) part += T[m][grp*16 + c] * awr[c];
    float g = sigmoid_f(part + __shfl_xor(part, 16, 32) + abv);
    if (!valid) g = 0.0f;                 // kill pad edge
    sG[m] = g;                            // unconditional, identical in halves
    __asm__ volatile("s_wait_dscnt 0" ::: "memory");

    // apply gate, accumulate m_i column sums, stage gated m for c-MLP
#pragma unroll
    for (int v = 0; v < 8; ++v) {
      const float gg = sG[mbase + v];
      const float q0 = s0[v] * gg;
      const float q1 = s1[v] * gg;
      T[mbase+v][n0] = q0; T[mbase+v][n0 + 16] = q1;
      cs0 += q0; cs1 += q1;
    }
    __asm__ volatile("s_wait_dscnt 0" ::: "memory");

    // coordinate MLP
    const v16h a3 = afrag_from_T(T, m, kh);
    v8f c0 = {}, c1 = {};
    c0 = wmma_f16(a3, Bc1_0, c0);
    c1 = wmma_f16(a3, Bc1_1, c1);
#pragma unroll
    for (int v = 0; v < 8; ++v) {
      T[mbase+v][n0]      = silu_f(c0[v] + cb1a);
      T[mbase+v][n0 + 16] = silu_f(c1[v] + cb1b);
    }
    __asm__ volatile("s_wait_dscnt 0" ::: "memory");

    float p2 = 0.0f;
#pragma unroll
    for (int c = 0; c < 16; ++c) p2 += T[m][grp*16 + c] * c2r[c];
    const float trans = tanhf(p2 + __shfl_xor(p2, 16, 32));
    const float coef  = valid ? (trans / (d + 1.0f)) : 0.0f;
    ux += r0 * coef; uy += r1 * coef; uz += r2 * coef;
  }

  // m_i: finish 16-row reduction, one store per lane (no atomics)
  cs0 += __shfl_xor(cs0, 16, 32);
  cs1 += __shfl_xor(cs1, 16, 32);
  mi_ws[(b * N_PART + i) * F_DIM + lane] = grp ? cs1 : cs0;

  // x_upd: reduce lanes 0..15 (lanes 16..31 are duplicates, masks stay <16)
#pragma unroll
  for (int mk = 1; mk < 16; mk <<= 1) {
    ux += __shfl_xor(ux, mk, 32);
    uy += __shfl_xor(uy, mk, 32);
    uz += __shfl_xor(uz, mk, 32);
  }
  if (lane == 0) {
    float* xo = x_out + (b * N_PART + i) * 3;
    xo[0] = xi0 + ux * 5.0f;   // COORDS_RANGE
    xo[1] = xi1 + uy * 5.0f;
    xo[2] = xi2 + uz * 5.0f;
  }
}

// ---------------------------------------------------------------------------
// Kernel 2: node MLP, one wave per block, 4 node-tiles per wave (256 blocks).
// h_out = h + silu([h | m_i] @ n_w1 + n_b1) @ n_w2 + n_b2
// ---------------------------------------------------------------------------
__global__ __launch_bounds__(32) void eq_gnn_node_kernel(
    const float* __restrict__ h,  const _Float16* __restrict__ h16,
    const float* __restrict__ mi,
    const float* __restrict__ nb1, const float* __restrict__ nb2,
    const _Float16* __restrict__ wp, float* __restrict__ h_out)
{
  __shared__ float T[16][32];
  const int lane  = threadIdx.x;
  const int m     = lane & 15;
  const int grp   = lane >> 4;
  const int mbase = grp * 8;
  const int kh    = grp * 8;
  const int n0    = m;

  const v16h Bn1_00 = load_frag(wp,  8, lane);
  const v16h Bn1_10 = load_frag(wp,  9, lane);
  const v16h Bn1_01 = load_frag(wp, 10, lane);
  const v16h Bn1_11 = load_frag(wp, 11, lane);
  const v16h Bn2_0  = load_frag(wp, 12, lane);
  const v16h Bn2_1  = load_frag(wp, 13, lane);
  const float b1a = nb1[n0], b1b = nb1[n0 + 16];
  const float b2a = nb2[n0], b2b = nb2[n0 + 16];

  for (int tt = 0; tt < 4; ++tt) {
    const int node0 = (blockIdx.x * 4 + tt) * 16;
    const int node  = node0 + m;

    const v16h a_h = afrag_f16row(h16 + node * F_DIM, kh);
    const float* mr = mi + node * F_DIM;
    v16h a_m;
#pragma unroll
    for (int q = 0; q < 8; ++q) {
      a_m[q]   = (_Float16)mr[kh + q];
      a_m[8+q] = (_Float16)mr[16 + kh + q];
    }

    v8f acc0 = {}, acc1 = {};
    acc0 = wmma_f16(a_h, Bn1_00, acc0);
    acc0 = wmma_f16(a_m, Bn1_10, acc0);
    acc1 = wmma_f16(a_h, Bn1_01, acc1);
    acc1 = wmma_f16(a_m, Bn1_11, acc1);
#pragma unroll
    for (int v = 0; v < 8; ++v) {
      T[mbase+v][n0]      = silu_f(acc0[v] + b1a);
      T[mbase+v][n0 + 16] = silu_f(acc1[v] + b1b);
    }
    __asm__ volatile("s_wait_dscnt 0" ::: "memory");

    const v16h a2 = afrag_from_T(T, m, kh);
    v8f o0 = {}, o1 = {};
    o0 = wmma_f16(a2, Bn2_0, o0);
    o1 = wmma_f16(a2, Bn2_1, o1);
#pragma unroll
    for (int v = 0; v < 8; ++v) {
      const int nm = node0 + mbase + v;
      h_out[nm * F_DIM + n0]      = h[nm * F_DIM + n0]      + o0[v] + b2a;
      h_out[nm * F_DIM + n0 + 16] = h[nm * F_DIM + n0 + 16] + o1[v] + b2b;
    }
    __asm__ volatile("s_wait_dscnt 0" ::: "memory");
  }
}

extern "C" void kernel_launch(void* const* d_in, const int* in_sizes, int n_in,
                              void* d_out, int out_size, void* d_ws, size_t ws_size,
                              hipStream_t stream) {
  (void)in_sizes; (void)n_in; (void)out_size; (void)ws_size;
  const float* x   = (const float*)d_in[0];
  const float* h   = (const float*)d_in[1];
  const float* dsb = (const float*)d_in[2];
  const float* ew1 = (const float*)d_in[3];
  const float* eb1 = (const float*)d_in[4];
  const float* ew2 = (const float*)d_in[5];
  const float* eb2 = (const float*)d_in[6];
  const float* nw1 = (const float*)d_in[7];
  const float* nb1 = (const float*)d_in[8];
  const float* nw2 = (const float*)d_in[9];
  const float* nb2 = (const float*)d_in[10];
  const float* cw1 = (const float*)d_in[11];
  const float* cb1 = (const float*)d_in[12];
  const float* cw2 = (const float*)d_in[13];
  const float* aw  = (const float*)d_in[14];
  const float* ab  = (const float*)d_in[15];

  float* out   = (float*)d_out;
  float* x_out = out;                       // (128,128,3)
  float* h_out = out + 128 * 128 * 3;       // (128,128,32)

  _Float16* wp    = (_Float16*)d_ws;
  _Float16* h16   = (_Float16*)((char*)d_ws + WP_BYTES);
  float*    mi_ws = (float*)((char*)d_ws + WP_BYTES + H16_BYTES);

  pack_weights_kernel<<<dim3(1), dim3(448), 0, stream>>>(ew1, ew2, cw1, nw1, nw2, wp);
  pack_h_kernel<<<dim3(512), dim3(256), 0, stream>>>(h, h16);
  eq_gnn_edge_kernel<<<dim3(128 * 128), dim3(32), 0, stream>>>(
      x, h16, dsb, ew1, eb1, eb2, cb1, cw2, aw, ab, wp, x_out, mi_ws);
  eq_gnn_node_kernel<<<dim3(256), dim3(32), 0, stream>>>(
      h, h16, mi_ws, nb1, nb2, wp, h_out);
}
